// MultiHeadAttention_81260781240616
// MI455X (gfx1250) — matrix operand order repaired and checked by reading, not compiled, not run on hardware
//
#include <hip/hip_runtime.h>

// ---------------- types ----------------
typedef __attribute__((ext_vector_type(16))) __bf16          v16bf;
typedef __attribute__((ext_vector_type(16))) unsigned short  v16u;
typedef __attribute__((ext_vector_type(8)))  unsigned short  v8u;
typedef __attribute__((ext_vector_type(8)))  float           v8f;

// ---------------- problem constants ----------------
constexpr int BB = 4;     // batch
constexpr int TT = 2048;  // sequence
constexpr int CE = 384;   // embed
constexpr int HH = 6;     // heads
constexpr int DH = 64;    // head size
constexpr int MTOK = BB * TT;  // 8192 tokens

// ---------------- CDNA5 async LDS copy (guarded) ----------------
#if defined(__has_builtin)
#if __has_builtin(__builtin_amdgcn_global_load_async_to_lds_b128) && \
    __has_builtin(__builtin_amdgcn_s_wait_asynccnt)
#define USE_ASYNC_LDS 1
#endif
#endif
#ifndef USE_ASYNC_LDS
#define USE_ASYNC_LDS 0
#endif

#if USE_ASYNC_LDS
typedef int v4i_vs __attribute__((vector_size(16)));
typedef __attribute__((address_space(1))) v4i_vs* g_v4i_ptr;
typedef __attribute__((address_space(3))) v4i_vs* l_v4i_ptr;
#endif

// Copy 16 bytes global -> LDS.  Async path uses ASYNCcnt-tracked
// global_load_async_to_lds_b128; fallback is a plain vector round trip.
__device__ __forceinline__ void cp16_g2l(unsigned short* dst_lds,
                                         const unsigned short* src_g) {
#if USE_ASYNC_LDS
  unsigned short* s = const_cast<unsigned short*>(src_g);
  __builtin_amdgcn_global_load_async_to_lds_b128((g_v4i_ptr)s, (l_v4i_ptr)dst_lds,
                                                 0, 0);
#else
  *(v8u*)dst_lds = *(const v8u*)src_g;
#endif
}

__device__ __forceinline__ void cp_wait() {
#if USE_ASYNC_LDS
  __builtin_amdgcn_s_wait_asynccnt(0);
#endif
}

// ---------------- helpers ----------------
__device__ __forceinline__ unsigned short f2bf(float f) {
  unsigned u = __float_as_uint(f);
  u += 0x7FFFu + ((u >> 16) & 1u);   // round-to-nearest-even
  return (unsigned short)(u >> 16);
}

__device__ __forceinline__ v16bf mk16(v8u lo, v8u hi) {
  v16u v = __builtin_shufflevector(lo, hi, 0, 1, 2, 3, 4, 5, 6, 7,
                                   8, 9, 10, 11, 12, 13, 14, 15);
  return __builtin_bit_cast(v16bf, v);
}

// A fragment, 16x32 bf16 (row-major source, row stride ld halves).
// lanes 0-15: halves 0-7 = K(k0+0..7),  halves 8-15 = K(k0+16..23)
// lanes16-31: halves 0-7 = K(k0+8..15), halves 8-15 = K(k0+24..31)
__device__ __forceinline__ v16bf loadA16x32(const unsigned short* src, int ld,
                                            int mbase, int k0) {
  int lane = threadIdx.x & 31;
  int sel = lane >> 4;
  const unsigned short* p = src + (mbase + (lane & 15)) * ld + k0 + sel * 8;
  v8u lo = *(const v8u*)p;
  v8u hi = *(const v8u*)(p + 16);
  return mk16(lo, hi);
}

// B fragment, 32x16 bf16. Source is B^T row-major: row n holds K contiguous.
// N = nbase + (lane&15); halves e<8 -> K = k0 + sel*8 + e, e>=8 -> K = k0 + 16 + sel*8 + (e-8).
__device__ __forceinline__ v16bf loadB32x16(const unsigned short* src, int ld,
                                            int nbase, int k0) {
  int lane = threadIdx.x & 31;
  int sel = lane >> 4;
  const unsigned short* p = src + (nbase + (lane & 15)) * ld + k0 + sel * 8;
  v8u lo = *(const v8u*)p;
  v8u hi = *(const v8u*)(p + 16);
  return mk16(lo, hi);
}

__device__ __forceinline__ v8f wmma_bf16(v16bf a, v16bf b, v8f c) {
  return __builtin_amdgcn_wmma_f32_16x16x32_bf16(false, a, false, b,
                                                 (short)0, c, false, false);
}

// ---------------- conversion kernels ----------------
__global__ __launch_bounds__(256) void k_cvt_x(const float* __restrict__ x,
                                               unsigned short* __restrict__ xb,
                                               int n) {
  int i = blockIdx.x * 256 + threadIdx.x;
  if (i < n) xb[i] = f2bf(x[i]);
}

// WtAll[w][n][k] = W_w[k][n]  (bf16 transposed weights; w: 0=Q,1=K,2=V,3=P)
__global__ __launch_bounds__(256) void k_cvt_w(const float* __restrict__ Wq,
                                               const float* __restrict__ Wk,
                                               const float* __restrict__ Wv,
                                               const float* __restrict__ Wp,
                                               unsigned short* __restrict__ WtAll) {
  int i = blockIdx.x * 256 + threadIdx.x;
  if (i >= 4 * CE * CE) return;
  int w = i / (CE * CE);
  int r = i % (CE * CE);
  int n = r / CE;
  int k = r % CE;
  const float* W = (w == 0) ? Wq : (w == 1) ? Wk : (w == 2) ? Wv : Wp;
  WtAll[(size_t)w * CE * CE + n * CE + k] = f2bf(W[k * CE + n]);
}

// ---------------- QKV projection GEMM ----------------
// grid: 128 mtiles * (3 weights * 6 ntiles); 256 thr = 8 waves
// Weight tile (64 x 384 bf16 = 48 KB, contiguous) staged in LDS via async copy.
// wave owns 16(M) x 32(N); Q,K -> [B,H,T,Dh]; V -> [B,H,Dh,T] (transposed)
__global__ __launch_bounds__(256) void k_qkv(const unsigned short* __restrict__ xb,
                                             const unsigned short* __restrict__ WtAll,
                                             unsigned short* __restrict__ Qb,
                                             unsigned short* __restrict__ Kb,
                                             unsigned short* __restrict__ Vt) {
  __shared__ __align__(16) unsigned short Wl[64 * CE];  // 48 KB
  int bid = blockIdx.x;
  int mt = bid / 18;
  int rem = bid % 18;
  int w = rem / 6;
  int nt = rem % 6;
  int tid = threadIdx.x;
  int wave = tid >> 5;
  int lane = tid & 31;
  int sel = lane >> 4;
  int mbase = mt * 64 + (wave & 3) * 16;
  int nloc = (wave >> 2) * 32;           // local N within the 64-wide tile
  int nbase = nt * 64 + nloc;            // global N

  // stage B^T tile: rows nt*64 .. nt*64+63 of Wt are contiguous
  const unsigned short* wsrc = WtAll + (size_t)w * CE * CE + (size_t)nt * 64 * CE;
  for (int i = tid; i < (64 * CE) / 8; i += 256) cp16_g2l(&Wl[i * 8], wsrc + i * 8);
  cp_wait();
  __syncthreads();

  v8f acc0 = {};
  v8f acc1 = {};
  for (int k0 = 0; k0 < CE; k0 += 32) {
    __builtin_prefetch(xb + (mbase + (lane & 15)) * CE + k0 + 32, 0, 3);
    v16bf a = loadA16x32(xb, CE, mbase, k0);
    v16bf b0 = loadB32x16(Wl, CE, nloc, k0);
    v16bf b1 = loadB32x16(Wl, CE, nloc + 16, k0);
    acc0 = wmma_bf16(a, b0, acc0);
    acc1 = wmma_bf16(a, b1, acc1);
  }
#pragma unroll
  for (int r = 0; r < 8; ++r) {
    int m = mbase + r + sel * 8;
    int bb = m >> 11;          // token -> batch
    int t = m & (TT - 1);      // token -> position
#pragma unroll
    for (int j = 0; j < 2; ++j) {
      int n = nbase + j * 16 + (lane & 15);
      int h = n >> 6;
      int dh = n & 63;
      unsigned short bv = f2bf((j == 0) ? acc0[r] : acc1[r]);
      if (w == 0)
        Qb[((size_t)(bb * HH + h) * TT + t) * DH + dh] = bv;
      else if (w == 1)
        Kb[((size_t)(bb * HH + h) * TT + t) * DH + dh] = bv;
      else
        Vt[((size_t)(bb * HH + h) * DH + dh) * TT + t] = bv;
    }
  }
}

// ---------------- causal flash attention ----------------
// grid: B*H*32 blocks, 128 thr = 4 waves; each wave owns 16 q-rows.
// K tile and V tile staged in LDS per k-step via async copies (shared by all
// 4 waves); P round-trips through a per-wave LDS tile for layout conversion.
__global__ __launch_bounds__(128) void k_attn(const unsigned short* __restrict__ Qb,
                                              const unsigned short* __restrict__ Kb,
                                              const unsigned short* __restrict__ Vt,
                                              unsigned short* __restrict__ Ob) {
  __shared__ __align__(16) unsigned short Kl[64 * 64];      // 8 KB
  __shared__ __align__(16) unsigned short Vl[64 * 64];      // 8 KB
  __shared__ __align__(16) unsigned short Pl[4][16 * 64];   // 8 KB
  int qt = blockIdx.x & 31;
  int bh = blockIdx.x >> 5;
  int b = bh / HH;
  int h = bh % HH;
  int tid = threadIdx.x;
  int wave = tid >> 5;
  int lane = tid & 31;
  int sel = lane >> 4;
  const unsigned short* Qh = Qb + (size_t)bh * TT * DH;
  const unsigned short* Kh = Kb + (size_t)bh * TT * DH;
  const unsigned short* Vh = Vt + (size_t)bh * DH * TT;
  int mbase = qt * 64 + wave * 16;

  v8f o[4];
  v8f zv = {};
#pragma unroll
  for (int t = 0; t < 4; ++t) o[t] = zv;
  float mrow[8], lrow[8];
#pragma unroll
  for (int r = 0; r < 8; ++r) { mrow[r] = -3.0e38f; lrow[r] = 0.f; }

  for (int kt = 0; kt <= qt; ++kt) {
    // ---- stage K (contiguous 8 KB) and V (64 rows x 128 B) tiles ----
    const unsigned short* ks = Kh + (size_t)kt * 64 * DH;
    for (int i = tid; i < 512; i += 128) cp16_g2l(&Kl[i * 8], ks + i * 8);
    for (int i = tid; i < 512; i += 128) {
      int r = i >> 3;
      int c = (i & 7) * 8;
      cp16_g2l(&Vl[r * 64 + c], Vh + (size_t)r * TT + kt * 64 + c);
    }
    cp_wait();
    __syncthreads();

    // ---- S = (Q K^T) * 1/sqrt(Dh), 16x64 tile per wave (4 N-tiles) ----
    v8f s[4];
#pragma unroll
    for (int j = 0; j < 4; ++j) s[j] = zv;
#pragma unroll
    for (int kc = 0; kc < DH; kc += 32) {
      v16bf a = loadA16x32(Qh, DH, mbase, kc);
#pragma unroll
      for (int j = 0; j < 4; ++j) {
        v16bf bk = loadB32x16(Kl, DH, j * 16, kc);
        s[j] = wmma_bf16(a, bk, s[j]);
      }
    }
    // ---- scale + causal mask (only diagonal k-tile needs masking) ----
#pragma unroll
    for (int j = 0; j < 4; ++j) {
#pragma unroll
      for (int r = 0; r < 8; ++r) {
        float vv = s[j][r] * 0.125f;  // 1/sqrt(64)
        if (kt == qt) {
          int key = kt * 64 + j * 16 + (lane & 15);
          int qr = mbase + r + sel * 8;
          if (key > qr) vv = -1.0e30f;
        }
        s[j][r] = vv;
      }
    }
    // ---- online softmax: row max over 64 keys ----
    float nm[8];
#pragma unroll
    for (int r = 0; r < 8; ++r) {
      float v = fmaxf(fmaxf(s[0][r], s[1][r]), fmaxf(s[2][r], s[3][r]));
      v = fmaxf(v, mrow[r]);
      v = fmaxf(v, __shfl_xor(v, 1, 32));
      v = fmaxf(v, __shfl_xor(v, 2, 32));
      v = fmaxf(v, __shfl_xor(v, 4, 32));
      v = fmaxf(v, __shfl_xor(v, 8, 32));
      nm[r] = v;
    }
    float rs[8];
#pragma unroll
    for (int r = 0; r < 8; ++r) rs[r] = 0.f;
#pragma unroll
    for (int j = 0; j < 4; ++j) {
#pragma unroll
      for (int r = 0; r < 8; ++r) {
        float p = __expf(s[j][r] - nm[r]);
        s[j][r] = p;
        rs[r] += p;
      }
    }
#pragma unroll
    for (int r = 0; r < 8; ++r) {
      float v = rs[r];
      v += __shfl_xor(v, 1, 32);
      v += __shfl_xor(v, 2, 32);
      v += __shfl_xor(v, 4, 32);
      v += __shfl_xor(v, 8, 32);
      float alpha = __expf(mrow[r] - nm[r]);
      lrow[r] = lrow[r] * alpha + v;
      mrow[r] = nm[r];
#pragma unroll
      for (int t = 0; t < 4; ++t) o[t][r] = o[t][r] * alpha;
    }
    // ---- stage P through LDS: C-layout -> A-layout ----
    unsigned short* pw = &Pl[wave][0];
#pragma unroll
    for (int j = 0; j < 4; ++j) {
#pragma unroll
      for (int r = 0; r < 8; ++r) {
        pw[(r + sel * 8) * 64 + j * 16 + (lane & 15)] = f2bf(s[j][r]);
      }
    }
    __syncthreads();
    // ---- O += P @ V (V tile in LDS, dh-major) ----
#pragma unroll
    for (int kc = 0; kc < 64; kc += 32) {
      v16bf a = loadA16x32(pw, 64, 0, kc);
#pragma unroll
      for (int t = 0; t < 4; ++t) {
        v16bf bv = loadB32x16(Vl, 64, t * 16, kc);
        o[t] = wmma_bf16(a, bv, o[t]);
      }
    }
    __syncthreads();   // all waves done reading Kl/Vl before next stage
  }
  // ---- epilogue: O /= l, store [B,T,H*Dh] bf16 ----
#pragma unroll
  for (int r = 0; r < 8; ++r) {
    float inv = 1.0f / lrow[r];
    int tok = mbase + r + sel * 8;
    size_t row = (size_t)b * TT + tok;
#pragma unroll
    for (int t = 0; t < 4; ++t) {
      int col = h * DH + t * 16 + (lane & 15);
      Ob[row * CE + col] = f2bf(o[t][r] * inv);
    }
  }
}

// ---------------- output projection GEMM + bias ----------------
__global__ __launch_bounds__(256) void k_proj(const unsigned short* __restrict__ Ob,
                                              const unsigned short* __restrict__ WpT,
                                              const float* __restrict__ bp,
                                              float* __restrict__ out) {
  __shared__ __align__(16) unsigned short Wl[64 * CE];  // 48 KB
  int bid = blockIdx.x;
  int mt = bid / 6;
  int nt = bid % 6;
  int tid = threadIdx.x;
  int wave = tid >> 5;
  int lane = tid & 31;
  int sel = lane >> 4;
  int mbase = mt * 64 + (wave & 3) * 16;
  int nloc = (wave >> 2) * 32;
  int nbase = nt * 64 + nloc;

  const unsigned short* wsrc = WpT + (size_t)nt * 64 * CE;
  for (int i = tid; i < (64 * CE) / 8; i += 256) cp16_g2l(&Wl[i * 8], wsrc + i * 8);
  cp_wait();
  __syncthreads();

  v8f acc0 = {};
  v8f acc1 = {};
  for (int k0 = 0; k0 < CE; k0 += 32) {
    __builtin_prefetch(Ob + (mbase + (lane & 15)) * CE + k0 + 32, 0, 3);
    v16bf a = loadA16x32(Ob, CE, mbase, k0);
    v16bf b0 = loadB32x16(Wl, CE, nloc, k0);
    v16bf b1 = loadB32x16(Wl, CE, nloc + 16, k0);
    acc0 = wmma_bf16(a, b0, acc0);
    acc1 = wmma_bf16(a, b1, acc1);
  }
#pragma unroll
  for (int r = 0; r < 8; ++r) {
    int m = mbase + r + sel * 8;
#pragma unroll
    for (int j = 0; j < 2; ++j) {
      int n = nbase + j * 16 + (lane & 15);
      out[(size_t)m * CE + n] = ((j == 0) ? acc0[r] : acc1[r]) + bp[n];
    }
  }
}

// ---------------- host launcher ----------------
extern "C" void kernel_launch(void* const* d_in, const int* in_sizes, int n_in,
                              void* d_out, int out_size, void* d_ws, size_t ws_size,
                              hipStream_t stream) {
  const float* x  = (const float*)d_in[0];
  const float* Wq = (const float*)d_in[1];
  const float* Wk = (const float*)d_in[2];
  const float* Wv = (const float*)d_in[3];
  const float* Wp = (const float*)d_in[4];
  const float* bp = (const float*)d_in[5];
  float* out = (float*)d_out;

  const size_t szX = (size_t)MTOK * CE * sizeof(unsigned short);     // 6.29 MB
  const size_t szW = (size_t)4 * CE * CE * sizeof(unsigned short);   // 1.18 MB
  const size_t szQ = (size_t)BB * HH * TT * DH * sizeof(unsigned short);  // 6.29 MB

  char* base = (char*)d_ws;
  unsigned short* xb    = (unsigned short*)(base);
  unsigned short* WtAll = (unsigned short*)(base + szX);
  unsigned short* Qb    = (unsigned short*)(base + szX + szW);
  unsigned short* Kb    = (unsigned short*)(base + szX + szW + szQ);
  unsigned short* Vt    = (unsigned short*)(base + szX + szW + 2 * szQ);
  unsigned short* Ob    = (unsigned short*)(base + szX + szW + 3 * szQ);

  const unsigned short* WpT = WtAll + (size_t)3 * CE * CE;

  // 1) convert inputs to bf16 (weights transposed)
  int nx = MTOK * CE;
  k_cvt_x<<<(nx + 255) / 256, 256, 0, stream>>>(x, xb, nx);
  k_cvt_w<<<(4 * CE * CE) / 256, 256, 0, stream>>>(Wq, Wk, Wv, Wp, WtAll);

  // 2) QKV projections (V stored transposed per head)
  k_qkv<<<128 * 18, 256, 0, stream>>>(xb, WtAll, Qb, Kb, Vt);

  // 3) causal flash attention
  k_attn<<<BB * HH * 32, 128, 0, stream>>>(Qb, Kb, Vt, Ob);

  // 4) output projection + bias (fp32 out)
  k_proj<<<128 * 6, 256, 0, stream>>>(Ob, WpT, bp, out);
}
